// SSDLayerLoss_27273042329736
// MI455X (gfx1250) — compile-verified, matrix operand order checked by reading, not run on hardware
//
#include <hip/hip_runtime.h>
#include <hip/hip_bf16.h>
#include <math.h>
#include <stdint.h>

// SSD multibox loss for MI455X (gfx1250, wave32).
// Bandwidth-bound: dominant cost is ONE streaming pass over conf_p (90.5 MB
// @ 23.3 TB/s ~= 4us). lc and ce are numerically identical (stop_gradient is
// grad-only), so we compute both in a single pass with a wave-per-row
// logsumexp (324 contiguous bytes per wave -> coalesced). Priors (140 KB)
// are served from the 192 MB L2. Block-level float reductions go through
// V_WMMA_F32_16X16X4_F32 ("multiply by ones" reduction on the matrix pipe).
// The negative-mining kernel stages its 35 KB lc slice into LDS with the
// Tensor Data Mover (tensor_load_to_lds + s_wait_tensorcnt) when available.

#define SSD_C   81
#define SSD_NB  32
#define SSD_NP  8732
#define SSD_NM  50
#define THREADS 256

typedef __attribute__((ext_vector_type(2)))  float    v2f;
typedef __attribute__((ext_vector_type(8)))  float    v8f;
typedef __attribute__((ext_vector_type(16))) _Float16 v16h;
typedef __attribute__((ext_vector_type(4)))  unsigned u32x4;
typedef __attribute__((ext_vector_type(4)))  int      i32x4;
typedef __attribute__((ext_vector_type(8)))  int      i32x8;

#if __has_builtin(__builtin_amdgcn_tensor_load_to_lds) && \
    __has_builtin(__builtin_amdgcn_s_wait_tensorcnt)
#define SSD_HAS_TDM 1
#else
#define SSD_HAS_TDM 0
#endif

// Reduce 256 per-thread floats to one sum (valid on wave 0 / thread 0).
// Uses WMMA with an all-ones A matrix: D = ones(16x4) x B(4x16) + C, so
// sum(all D elements) == 16 * sum(B). Four accumulating WMMAs consume all
// 256 partials; a 32-lane shuffle finishes. EXEC is all-ones within wave 0.
__device__ inline float block_sum_256(float v, float* sm) {
  sm[threadIdx.x] = v;
  __syncthreads();
  float total = 0.0f;
  if (threadIdx.x < 32) {
    const int lane = threadIdx.x;
    v8f c = {};
#if __has_builtin(__builtin_amdgcn_wmma_f32_16x16x4_f32)
    v2f ones; ones.x = 1.0f; ones.y = 1.0f;
    for (int g = 0; g < 4; ++g) {
      v2f b; b.x = sm[g * 64 + lane]; b.y = sm[g * 64 + 32 + lane];
      c = __builtin_amdgcn_wmma_f32_16x16x4_f32(false, ones, false, b,
                                                (short)0, c, false, false);
    }
#else
    // Fallback (probe-confirmed builtin): f16 ones x f16 partials.
    v16h ones, b;
    for (int j = 0; j < 16; ++j) ones[j] = (_Float16)1.0f;
    for (int j = 0; j < 16; ++j)
      b[j] = (j < 8) ? (_Float16)sm[lane * 8 + j] : (_Float16)0.0f;
    c = __builtin_amdgcn_wmma_f32_16x16x32_f16(false, ones, false, b,
                                               (short)0, c, false, false);
#endif
    float s = c[0] + c[1] + c[2] + c[3] + c[4] + c[5] + c[6] + c[7];
    for (int off = 16; off > 0; off >>= 1) s += __shfl_xor(s, off, 32);
    total = s * (1.0f / 16.0f);
  }
  __syncthreads();
  return total;
}

// ---------------------------------------------------------------- kernel A
// One block per batch item: bidirectional matching, encode, smooth-L1.
__global__ __launch_bounds__(THREADS) void ssd_match_kernel(
    const float* __restrict__ loc_p, const float* __restrict__ priorbox,
    const float* __restrict__ targets, int* __restrict__ conf_lab,
    float* __restrict__ loss_acc /*[0]=loc,[1]=conf*/,
    int* __restrict__ num_pos_total, int* __restrict__ num_pos_batch) {
  __shared__ float s_box[SSD_NM][4];
  __shared__ float s_area[SSD_NM];
  __shared__ float s_label[SSD_NM];
  __shared__ int   s_valid[SSD_NM];
  __shared__ unsigned long long s_best[SSD_NM];  // (iou_bits<<32) | ~prior
  __shared__ int   s_forced_p[SSD_NM];
  __shared__ float s_red[THREADS];
  __shared__ int   s_pos_cnt;

  const int n = blockIdx.x;
  const int t = threadIdx.x;
  if (t == 0) s_pos_cnt = 0;
  if (t < SSD_NM) {
    const float* tg = targets + ((size_t)n * SSD_NM + t) * 5;
    float lab = tg[0];
    float x0 = tg[1], y0 = tg[2], x1 = tg[3], y1 = tg[4];
    s_label[t] = lab;
    s_box[t][0] = x0; s_box[t][1] = y0; s_box[t][2] = x1; s_box[t][3] = y1;
    s_area[t]  = (x1 - x0) * (y1 - y0);
    s_valid[t] = lab > 0.0f;
    s_best[t]  = 0ull;
    s_forced_p[t] = -1;
  }
  __syncthreads();

  // Per-GT-box best prior (argmax over P). Packed key: higher IoU wins,
  // ties -> larger ~p -> lower prior index (matches jnp.argmax).
  for (int m = 0; m < SSD_NM; ++m) {
    if (!s_valid[m]) continue;
    const float ax0 = s_box[m][0], ay0 = s_box[m][1];
    const float ax1 = s_box[m][2], ay1 = s_box[m][3];
    const float aa = s_area[m];
    unsigned long long lk = 0ull;
    for (int p = t; p < SSD_NP; p += THREADS) {
      const float* pb = priorbox + (size_t)p * 4;
      float bx0 = pb[0], by0 = pb[1], bx1 = pb[2], by1 = pb[3];
      float iw = fmaxf(fminf(ax1, bx1) - fmaxf(ax0, bx0), 0.0f);
      float ih = fmaxf(fminf(ay1, by1) - fmaxf(ay0, by0), 0.0f);
      float inter = iw * ih;
      float iou = inter / (aa + (bx1 - bx0) * (by1 - by0) - inter);
      unsigned long long key =
          ((unsigned long long)__float_as_uint(iou) << 32) |
          (unsigned)(~(unsigned)p);
      lk = key > lk ? key : lk;
    }
    atomicMax(&s_best[m], lk);
  }
  __syncthreads();
  if (t < SSD_NM && s_valid[t])
    s_forced_p[t] = (int)(~(unsigned)(s_best[t] & 0xFFFFFFFFull));
  __syncthreads();

  // Per prior: best GT (first-max), forced override, label, encode, SL1.
  float lloss = 0.0f;
  int   lpos  = 0;
  for (int p = t; p < SSD_NP; p += THREADS) {
    const float* pb = priorbox + (size_t)p * 4;
    float bx0 = pb[0], by0 = pb[1], bx1 = pb[2], by1 = pb[3];
    float pw = bx1 - bx0, ph = by1 - by0;
    float pcx = (bx0 + bx1) * 0.5f, pcy = (by0 + by1) * 0.5f;
    float ab = pw * ph;
    float best = -1e30f; int bidx = 0;
    for (int m = 0; m < SSD_NM; ++m) {
      float v;
      if (s_valid[m]) {
        float iw = fmaxf(fminf(s_box[m][2], bx1) - fmaxf(s_box[m][0], bx0), 0.0f);
        float ih = fmaxf(fminf(s_box[m][3], by1) - fmaxf(s_box[m][1], by0), 0.0f);
        float inter = iw * ih;
        v = inter / (s_area[m] + ab - inter);
      } else {
        v = -1.0f;
      }
      if (v > best) { best = v; bidx = m; }
    }
    int forced = 0, fidx = 0;
    for (int m = 0; m < SSD_NM; ++m)
      if (s_forced_p[m] == p) { forced = 1; fidx = m; }  // last hit = max m
    const int gt = forced ? fidx : bidx;
    float conf = s_label[gt];
    if (best < 0.5f && !forced) conf = 0.0f;
    const int lab = (int)conf;
    conf_lab[(size_t)n * SSD_NP + p] = lab;
    if (lab > 0) {
      ++lpos;
      float mx0 = s_box[gt][0], my0 = s_box[gt][1];
      float mx1 = s_box[gt][2], my1 = s_box[gt][3];
      float g4[4];
      g4[0] = ((mx0 + mx1) * 0.5f - pcx) / (0.1f * pw);
      g4[1] = ((my0 + my1) * 0.5f - pcy) / (0.1f * ph);
      g4[2] = logf(fmaxf((mx1 - mx0) / pw, 1e-6f)) * 5.0f;  // /0.2
      g4[3] = logf(fmaxf((my1 - my0) / ph, 1e-6f)) * 5.0f;
      const float* lp = loc_p + ((size_t)n * SSD_NP + p) * 4;
      for (int k2 = 0; k2 < 4; ++k2) {
        float d = lp[k2] - g4[k2];
        float ad = fabsf(d);
        lloss += (ad < 1.0f) ? 0.5f * d * d : ad - 0.5f;
      }
    }
  }
  atomicAdd(&s_pos_cnt, lpos);
  float bsum = block_sum_256(lloss, s_red);  // contains __syncthreads
  if (t == 0) {
    atomicAdd(&loss_acc[0], bsum);
    atomicAdd(num_pos_total, s_pos_cnt);
    num_pos_batch[n] = s_pos_cnt;
  }
}

// ---------------------------------------------------------------- kernel B
// Wave-per-row logsumexp over 81 classes; single pass over conf_p.
__global__ __launch_bounds__(THREADS) void ssd_conf_kernel(
    const float* __restrict__ conf_p, const int* __restrict__ conf_lab,
    float* __restrict__ lc_out, float* __restrict__ ce_out) {
  const int wid  = threadIdx.x >> 5;
  const int lane = threadIdx.x & 31;
  const long long row = (long long)blockIdx.x * 8 + wid;
  if (row >= (long long)SSD_NB * SSD_NP) return;
  const float* cp = conf_p + row * SSD_C;
  __builtin_prefetch(cp + 8 * SSD_C, 0, 3);  // next block's rows -> global_prefetch_b8
  float x0 = cp[lane];
  float x1 = cp[lane + 32];
  float x2 = (lane + 64 < SSD_C) ? cp[lane + 64] : -3.402823466e38f;
  float mx = fmaxf(fmaxf(x0, x1), x2);
  for (int off = 16; off > 0; off >>= 1) mx = fmaxf(mx, __shfl_xor(mx, off, 32));
  float s = __expf(x0 - mx) + __expf(x1 - mx) +
            ((lane + 64 < SSD_C) ? __expf(x2 - mx) : 0.0f);
  for (int off = 16; off > 0; off >>= 1) s += __shfl_xor(s, off, 32);
  if (lane == 0) {
    float lse = mx + __logf(s);
    int lab = conf_lab[row];
    float ce = lse - cp[lab];
    ce_out[row] = ce;
    lc_out[row] = (lab > 0) ? 0.0f : ce;  // pos entries excluded from ranking
  }
}

// ---------------------------------------------------------------- kernel C
// One block per batch: k-th-largest selection of lc on float bits (lc >= 0,
// so uint order == float order), stable tie handling, then sum ce*(pos|neg).
// The lc slice (8732 floats = 34928 B, re-read ~35x by the bisection) is
// staged into LDS by the Tensor Data Mover: one 2D D# (8732x1 x 4B tile),
// issued by wave 0, completion on TENSORcnt.
__global__ __launch_bounds__(THREADS) void ssd_neg_kernel(
    const float* __restrict__ lc, const float* __restrict__ ce,
    const int* __restrict__ conf_lab, const int* __restrict__ num_pos_batch,
    float* __restrict__ loss_acc) {
  __shared__ float s_lc[SSD_NP];
  __shared__ float s_red[THREADS];
  __shared__ int   s_cnt;
  __shared__ unsigned s_vk;
  __shared__ int   s_tcut;

  const int n = blockIdx.x;
  const int t = threadIdx.x;
  const size_t base = (size_t)n * SSD_NP;

#if SSD_HAS_TDM
  if (t < 32) {  // wave 0 issues the DMA (TDM ignores EXEC; one issue/wave)
    const unsigned lds_off = (unsigned)(uintptr_t)(&s_lc[0]);
    const unsigned long long ga = (unsigned long long)(uintptr_t)(lc + base);
    u32x4 g0;
    g0[0] = 1u;                                   // count=1, user descriptor
    g0[1] = lds_off;                              // lds_addr [63:32]
    g0[2] = (unsigned)(ga & 0xFFFFFFFFull);       // global_addr[31:0]
    g0[3] = (unsigned)((ga >> 32) & 0x01FFFFFFull) | 0x80000000u;  // type=2
    i32x8 g1;
    g1[0] = 0x00020000;                           // data_size=4B, no multicast
    g1[1] = (int)(((unsigned)SSD_NP & 0xFFFFu) << 16);  // tensor_dim0 lo16
    g1[2] = (int)((((unsigned)SSD_NP >> 16) & 0xFFFFu) | (1u << 16)); // dim0 hi / tensor_dim1=1
    g1[3] = (int)(((unsigned)SSD_NP & 0xFFFFu) << 16);  // tile_dim0=8732
    g1[4] = 1;                                    // tile_dim1=1, tile_dim2=0
    g1[5] = SSD_NP;                               // tensor_dim0_stride lo32
    g1[6] = 0;                                    // stride hi, dim1_stride lo
    g1[7] = 0;
    i32x4 gz; gz[0] = 0; gz[1] = 0; gz[2] = 0; gz[3] = 0;
#if __clang_major__ >= 23
    i32x8 gz8;
    for (int j = 0; j < 8; ++j) gz8[j] = 0;
    __builtin_amdgcn_tensor_load_to_lds(g0, g1, gz, gz, gz8, 0);
#else
    __builtin_amdgcn_tensor_load_to_lds(g0, g1, gz, gz, 0);
#endif
    __builtin_amdgcn_s_wait_tensorcnt(0);
  }
#else
  for (int i = t; i < SSD_NP; i += THREADS) s_lc[i] = lc[base + i];
#endif
  int np = num_pos_batch[n];
  int k = 3 * np;
  if (k > SSD_NP - 1) k = SSD_NP - 1;
  __syncthreads();  // TDM data (or manual copy) visible to all waves

  unsigned vk; int tcut;
  if (k <= 0) {  // uniform branch per block
    vk = 0xFFFFFFFFu; tcut = -1;
  } else {
    unsigned lo = 0u, hi = 0x7F800000u;  // bisection: largest x with cnt_ge>=k
    while (lo < hi) {
      unsigned mid = lo + ((hi - lo + 1u) >> 1);
      if (t == 0) s_cnt = 0;
      __syncthreads();
      int lcount = 0;
      for (int i = t; i < SSD_NP; i += THREADS)
        if (__float_as_uint(s_lc[i]) >= mid) ++lcount;
      atomicAdd(&s_cnt, lcount);
      __syncthreads();
      int c = s_cnt;
      __syncthreads();
      if (c >= k) lo = mid; else hi = mid - 1u;
    }
    vk = lo;
    if (t == 0) s_cnt = 0;
    __syncthreads();
    int lcount = 0;
    for (int i = t; i < SSD_NP; i += THREADS)
      if (__float_as_uint(s_lc[i]) > vk) ++lcount;
    atomicAdd(&s_cnt, lcount);
    __syncthreads();
    if (t == 0) {
      int r = k - s_cnt;  // ties to take, in ascending index (stable sort)
      int cut = -1, taken = 0;
      for (int i = 0; i < SSD_NP && taken < r; ++i)
        if (__float_as_uint(s_lc[i]) == vk) { ++taken; cut = i; }
      s_tcut = cut; s_vk = vk;
    }
    __syncthreads();
    vk = s_vk; tcut = s_tcut;
  }

  float lsum = 0.0f;
  for (int i = t; i < SSD_NP; i += THREADS) {
    int lab = conf_lab[base + i];
    unsigned b = __float_as_uint(s_lc[i]);
    int neg = (b > vk) || (b == vk && (int)i <= tcut);
    if (lab > 0 || neg) lsum += ce[base + i];
  }
  float bsum = block_sum_256(lsum, s_red);
  if (t == 0) atomicAdd(&loss_acc[1], bsum);
}

// ------------------------------------------------------------- init / final
__global__ void ssd_init_kernel(float* acc, int* np_total) {
  if (threadIdx.x == 0) { acc[0] = 0.0f; acc[1] = 0.0f; *np_total = 0; }
}
__global__ void ssd_final_kernel(const float* acc, const int* np_total,
                                 float* out) {
  if (threadIdx.x == 0) out[0] = (acc[0] + acc[1]) / (float)(*np_total);
}

// ------------------------------------------------------------------ launch
extern "C" void kernel_launch(void* const* d_in, const int* in_sizes, int n_in,
                              void* d_out, int out_size, void* d_ws,
                              size_t ws_size, hipStream_t stream) {
  const float* loc_p    = (const float*)d_in[0];  // (32, 8732, 4)
  const float* conf_p   = (const float*)d_in[1];  // (32, 8732, 81)
  const float* priorbox = (const float*)d_in[2];  // (8732, 4)
  const float* targets  = (const float*)d_in[3];  // (32, 50, 5)
  float* out = (float*)d_out;

  char* ws = (char*)d_ws;
  int*   conf_lab = (int*)ws;            ws += (size_t)SSD_NB * SSD_NP * sizeof(int);
  float* lc       = (float*)ws;          ws += (size_t)SSD_NB * SSD_NP * sizeof(float);
  float* ce       = (float*)ws;          ws += (size_t)SSD_NB * SSD_NP * sizeof(float);
  int*   np_batch = (int*)ws;            ws += 64 * sizeof(int);
  float* acc      = (float*)ws;          ws += 2 * sizeof(float);
  int*   np_total = (int*)ws;

  ssd_init_kernel<<<1, 64, 0, stream>>>(acc, np_total);
  ssd_match_kernel<<<SSD_NB, THREADS, 0, stream>>>(
      loc_p, priorbox, targets, conf_lab, acc, np_total, np_batch);
  const long long rows = (long long)SSD_NB * SSD_NP;
  ssd_conf_kernel<<<(unsigned)((rows + 7) / 8), THREADS, 0, stream>>>(
      conf_p, conf_lab, lc, ce);
  ssd_neg_kernel<<<SSD_NB, THREADS, 0, stream>>>(lc, ce, conf_lab, np_batch, acc);
  ssd_final_kernel<<<1, 64, 0, stream>>>(acc, np_total, out);
  (void)in_sizes; (void)n_in; (void)out_size; (void)ws_size;
}
